// MultiheadAttention_59768764891555
// MI455X (gfx1250) — matrix operand order repaired and checked by reading, not compiled, not run on hardware
//
#include <hip/hip_runtime.h>
#include <hip/hip_bf16.h>

// ---------------------------------------------------------------------------
// Types
// ---------------------------------------------------------------------------
typedef __attribute__((ext_vector_type(16))) __bf16        v16bf;
typedef __attribute__((ext_vector_type(8)))  float         v8f;
typedef __attribute__((ext_vector_type(8)))  unsigned int  v8u;
typedef __attribute__((ext_vector_type(4)))  unsigned int  v4u;
typedef __attribute__((ext_vector_type(4)))  int           v4i;

typedef __attribute__((address_space(1))) v4i* glob_v4i_ptr;
typedef __attribute__((address_space(3))) v4i* lds_v4i_ptr;

// async 16B global -> LDS copy (per-lane addresses, one ASYNCcnt per call site)
__device__ __forceinline__ void async_cp16(const void* g, void* l) {
  __builtin_amdgcn_global_load_async_to_lds_b128((glob_v4i_ptr)g,
                                                 (lds_v4i_ptr)l, 0, 0);
}

// fp32 -> bf16, round-to-nearest-even
__device__ __forceinline__ unsigned short f2bf(float x) {
  unsigned u = __builtin_bit_cast(unsigned, x);
  unsigned r = u + 0x7FFFu + ((u >> 16) & 1u);
  return (unsigned short)(r >> 16);
}
__device__ __forceinline__ unsigned pk2(float a, float b) {
  return (unsigned)f2bf(a) | ((unsigned)f2bf(b) << 16);
}

// A-matrix fragment (16x32 bf16, M x K): lane holds row = lane&15,
// K = {h*8..h*8+7} U {16+h*8..16+h*8+7}, h = lane>>4.
__device__ __forceinline__ v16bf load_frag_a(const unsigned short* base,
                                             int strideElems, int lane) {
  const unsigned short* p = base + (lane & 15) * strideElems + ((lane >> 4) << 3);
  v4u lo = *(const v4u*)(p);
  v4u hi = *(const v4u*)(p + 16);
  v8u w;
  w[0] = lo[0]; w[1] = lo[1]; w[2] = lo[2]; w[3] = lo[3];
  w[4] = hi[0]; w[5] = hi[1]; w[6] = hi[2]; w[7] = hi[3];
  return __builtin_bit_cast(v16bf, w);
}

// B-matrix fragment (32x16 bf16, K x N): lane holds column N = lane&15,
// K = {h*8..h*8+7} U {16+h*8..16+h*8+7}, h = lane>>4. Data stored per-N-row, K contiguous.
__device__ __forceinline__ v16bf load_frag_b(const unsigned short* base,
                                             int strideElems, int lane) {
  const unsigned short* p = base + (lane & 15) * strideElems + ((lane >> 4) << 3);
  v4u lo = *(const v4u*)(p);
  v4u hi = *(const v4u*)(p + 16);
  v8u w;
  w[0] = lo[0]; w[1] = lo[1]; w[2] = lo[2]; w[3] = lo[3];
  w[4] = hi[0]; w[5] = hi[1]; w[6] = hi[2]; w[7] = hi[3];
  return __builtin_bit_cast(v16bf, w);
}

__device__ __forceinline__ v8f wmma_bf16(v16bf a, v16bf b, v8f c) {
  return __builtin_amdgcn_wmma_f32_16x16x32_bf16(false, a, false, b,
                                                 (short)0, c, false, false);
}

// ---------------------------------------------------------------------------
// fp32 -> bf16 bulk convert (for weights)
// ---------------------------------------------------------------------------
__global__ __launch_bounds__(256) void cvt_bf16(const float* __restrict__ in,
                                                unsigned short* __restrict__ out) {
  int i = (blockIdx.x * 256 + threadIdx.x) * 4;
  float4 d = *(const float4*)(in + i);
  uint2 p;
  p.x = pk2(d.x, d.y);
  p.y = pk2(d.z, d.w);
  *(uint2*)(out + i) = p;
}

// ---------------------------------------------------------------------------
// GEMM: Y = alpha * (X @ W^T)   X:[16384,1024] fp32 or bf16, W:[1024,1024] bf16
// Y bf16 or fp32. Block tile 128x128, K-step 32, 8 waves (4 M x 2 N).
// Double-buffered LDS; async-to-LDS pipelined one k-step ahead
// (wait asynccnt <= N_next ==> current tile landed; in-order completion).
// ---------------------------------------------------------------------------
template <bool XBF, bool OUTBF>
__global__ __launch_bounds__(256) void gemm_xwt(const void* __restrict__ Xv,
                                                const unsigned short* __restrict__ W,
                                                void* __restrict__ Yv,
                                                float alpha) {
  constexpr int KK = 1024, NN = 1024;
  constexpr int NK = KK / 32;  // 32 k-steps
  __shared__ unsigned short Xs[2][128 * 40];  // 32 K + 8 pad, rows 80B
  __shared__ unsigned short Ws[2][128 * 40];

  const int tid  = threadIdx.x;
  const int lane = tid & 31;
  const int wv   = tid >> 5;
  const int wm   = wv & 3;
  const int wn   = wv >> 2;
  const int m0   = blockIdx.y * 128;
  const int n0   = blockIdx.x * 128;

  v8f C[2][4];
#pragma unroll
  for (int i = 0; i < 2; ++i)
#pragma unroll
    for (int j = 0; j < 4; ++j)
#pragma unroll
      for (int e = 0; e < 8; ++e) C[i][j][e] = 0.0f;

  auto stage = [&](int kc, int pbuf) {
    const int k0 = kc * 32;
    // W tile (128 x 32 bf16): async copy (2 ASYNCcnt)
#pragma unroll
    for (int i = 0; i < 2; ++i) {
      int c = i * 256 + tid;
      int row = c >> 2, col8 = (c & 3) << 3;
      async_cp16(W + (size_t)(n0 + row) * KK + k0 + col8,
                 &Ws[pbuf][row * 40 + col8]);
    }
    if constexpr (XBF) {  // bf16 X: async copy (2 ASYNCcnt)
      const unsigned short* X = (const unsigned short*)Xv;
#pragma unroll
      for (int i = 0; i < 2; ++i) {
        int c = i * 256 + tid;
        int row = c >> 2, col8 = (c & 3) << 3;
        async_cp16(X + (size_t)(m0 + row) * KK + k0 + col8,
                   &Xs[pbuf][row * 40 + col8]);
      }
    } else {  // fp32 X: load + cvt + ds_store
      const float* X = (const float*)Xv;
#pragma unroll
      for (int i = 0; i < 4; ++i) {
        int c = i * 256 + tid;
        int row = c >> 3, col4 = (c & 7) << 2;
        float4 d = *(const float4*)(X + (size_t)(m0 + row) * KK + k0 + col4);
        uint2 p;
        p.x = pk2(d.x, d.y);
        p.y = pk2(d.z, d.w);
        *(uint2*)(&Xs[pbuf][row * 40 + col4]) = p;
      }
    }
  };
  constexpr int NASYNC = XBF ? 4 : 2;

  stage(0, 0);
  for (int kt = 0; kt < NK; ++kt) {
    const int buf = kt & 1;
    if (kt + 1 < NK) {
      stage(kt + 1, buf ^ 1);  // pipeline next tile into other buffer
      __builtin_amdgcn_s_wait_asynccnt(NASYNC);  // current tile has landed
    } else {
      __builtin_amdgcn_s_wait_asynccnt(0);
    }
    __syncthreads();

    v16bf a[2], bfr[4];
#pragma unroll
    for (int i = 0; i < 2; ++i)
      a[i] = load_frag_a(&Xs[buf][(wm * 32 + i * 16) * 40], 40, lane);
#pragma unroll
    for (int j = 0; j < 4; ++j)
      bfr[j] = load_frag_b(&Ws[buf][(wn * 64 + j * 16) * 40], 40, lane);
#pragma unroll
    for (int i = 0; i < 2; ++i)
#pragma unroll
      for (int j = 0; j < 4; ++j) C[i][j] = wmma_bf16(a[i], bfr[j], C[i][j]);
    __syncthreads();  // compute(buf) done before buf is restaged
  }

  const int rh = (lane >> 4) * 8;
  const int cn = lane & 15;
#pragma unroll
  for (int i = 0; i < 2; ++i)
#pragma unroll
    for (int j = 0; j < 4; ++j)
#pragma unroll
      for (int e = 0; e < 8; ++e) {
        int row = m0 + wm * 32 + i * 16 + e + rh;
        int col = n0 + wn * 64 + j * 16 + cn;
        float val = alpha * C[i][j][e];
        if constexpr (OUTBF)
          ((unsigned short*)Yv)[(size_t)row * NN + col] = f2bf(val);
        else
          ((float*)Yv)[(size_t)row * NN + col] = val;
      }
}

// ---------------------------------------------------------------------------
// Flash attention. One block = one (b, h, 128-query tile). 8 waves, each wave
// owns 16 query rows. Streams 64-key chunks, double-buffered:
//   K chunk  : async-to-LDS, issued one chunk ahead (wait asynccnt <= 2)
//   V chunk  : b128 global loads one chunk ahead into registers, LDS-transposed
//              scatter on the consuming iteration
// Q pre-scaled by 1/sqrt(DH). Output in head-interleaved "concat" layout.
// ---------------------------------------------------------------------------
__global__ __launch_bounds__(256) void attn_kernel(
    const unsigned short* __restrict__ Q, const unsigned short* __restrict__ Kb,
    const unsigned short* __restrict__ Vb, const int* __restrict__ mask,
    unsigned short* __restrict__ Cout) {
  constexpr int S = 1024, E = 1024;
  constexpr int NJ = S / 64;  // 16 key chunks
  __shared__ unsigned short Qs[128 * 72];       // [qrow][d], stride 144B
  __shared__ unsigned short Ks[2][64 * 72];     // [krow][d]
  __shared__ unsigned short Vt[2][64 * 72];     // [d][krow]
  __shared__ unsigned short Ps[8 * 16 * 72];    // per-wave P: [16 q][64 k]
  __shared__ float Msk[2][64];

  const int tid = threadIdx.x, lane = tid & 31, wv = tid >> 5;
  const int blk = blockIdx.x;
  const int qb = blk & 7, h = (blk >> 3) & 15, b = blk >> 7;
  const int q0 = qb * 128;
  const size_t base = ((size_t)b * S) * E + h * 64;

  // stage Q tile (128 x 64 bf16) via async copy
#pragma unroll
  for (int i = 0; i < 4; ++i) {
    int c = i * 256 + tid;
    int row = c >> 3, col8 = (c & 7) << 3;
    async_cp16(Q + base + (size_t)(q0 + row) * E + col8, &Qs[row * 72 + col8]);
  }
  __builtin_amdgcn_s_wait_asynccnt(0);
  __syncthreads();

  v16bf aq[2];
  aq[0] = load_frag_a(&Qs[(wv * 16) * 72 + 0], 72, lane);
  aq[1] = load_frag_a(&Qs[(wv * 16) * 72 + 32], 72, lane);

  float mst[8], lst[8];
  v8f O[4];
#pragma unroll
  for (int e = 0; e < 8; ++e) { mst[e] = -1e30f; lst[e] = 0.0f; }
#pragma unroll
  for (int td = 0; td < 4; ++td)
#pragma unroll
    for (int e = 0; e < 8; ++e) O[td][e] = 0.0f;

  const int cn = lane & 15;

  auto stageK = [&](int jc, int pbuf) {  // 2 ASYNCcnt
#pragma unroll
    for (int i = 0; i < 2; ++i) {
      int c = i * 256 + tid;
      int row = c >> 3, col8 = (c & 7) << 3;
      async_cp16(Kb + base + (size_t)(jc * 64 + row) * E + col8,
                 &Ks[pbuf][row * 72 + col8]);
    }
  };
  auto loadV = [&](int jc, v4u* r) {
#pragma unroll
    for (int i = 0; i < 2; ++i) {
      int task = i * 256 + tid;
      int kk = task >> 3, d8 = (task & 7) << 3;
      r[i] = *(const v4u*)(Vb + base + (size_t)(jc * 64 + kk) * E + d8);
    }
  };
  auto storeVt = [&](const v4u* r, int pbuf) {
#pragma unroll
    for (int i = 0; i < 2; ++i) {
      int task = i * 256 + tid;
      int kk = task >> 3, d8 = (task & 7) << 3;
#pragma unroll
      for (int e = 0; e < 8; ++e)
        Vt[pbuf][(d8 + e) * 72 + kk] =
            (unsigned short)(r[i][e >> 1] >> ((e & 1) * 16));
    }
  };
  auto stageM = [&](int jc, int pbuf) {
    if (tid < 64)
      Msk[pbuf][tid] = (mask[(size_t)b * S + jc * 64 + tid] == 0) ? 0.f : 1.f;
  };

  v4u vreg[2];
  loadV(0, vreg);
  stageK(0, 0);
  stageM(0, 0);

  for (int jc = 0; jc < NJ; ++jc) {
    const int buf = jc & 1;
    storeVt(vreg, buf);  // LDS-transpose scatter for the current chunk
    if (jc + 1 < NJ) {
      loadV(jc + 1, vreg);        // global latency hidden under compute
      stageK(jc + 1, buf ^ 1);    // async into the other buffer
      stageM(jc + 1, buf ^ 1);
      __builtin_amdgcn_s_wait_asynccnt(2);  // chunk jc's K has landed
    } else {
      __builtin_amdgcn_s_wait_asynccnt(0);
    }
    __syncthreads();

    // ---- logits: 16 x 64 per wave, contraction over d (2 x 32) ----
    v8f Lg[4];
#pragma unroll
    for (int kt = 0; kt < 4; ++kt) {
      v8f acc;
#pragma unroll
      for (int e = 0; e < 8; ++e) acc[e] = 0.0f;
      v16bf b0 = load_frag_b(&Ks[buf][(kt * 16) * 72 + 0], 72, lane);
      v16bf b1 = load_frag_b(&Ks[buf][(kt * 16) * 72 + 32], 72, lane);
      acc = wmma_bf16(aq[0], b0, acc);
      acc = wmma_bf16(aq[1], b1, acc);
      Lg[kt] = acc;
    }

    // ---- mask + online softmax stats ----
    float mx[8];
#pragma unroll
    for (int e = 0; e < 8; ++e) mx[e] = -1e30f;
#pragma unroll
    for (int kt = 0; kt < 4; ++kt) {
      float msel = Msk[buf][kt * 16 + cn];
#pragma unroll
      for (int e = 0; e < 8; ++e) {
        float lg = (msel != 0.f) ? Lg[kt][e] : -8964.0f;
        Lg[kt][e] = lg;
        mx[e] = fmaxf(mx[e], lg);
      }
    }
#pragma unroll
    for (int e = 0; e < 8; ++e) {
#pragma unroll
      for (int o = 1; o < 16; o <<= 1) mx[e] = fmaxf(mx[e], __shfl_xor(mx[e], o));
    }
    float sc[8];
#pragma unroll
    for (int e = 0; e < 8; ++e) {
      float mn = fmaxf(mst[e], mx[e]);
      sc[e] = __expf(mst[e] - mn);
      mst[e] = mn;
    }
    float sm[8];
#pragma unroll
    for (int e = 0; e < 8; ++e) sm[e] = 0.0f;
    const int rh = (lane >> 4) * 8;
#pragma unroll
    for (int kt = 0; kt < 4; ++kt) {
#pragma unroll
      for (int e = 0; e < 8; ++e) {
        float p = __expf(Lg[kt][e] - mst[e]);
        sm[e] += p;
        Ps[(wv * 16 + e + rh) * 72 + kt * 16 + cn] = f2bf(p);
      }
    }
#pragma unroll
    for (int e = 0; e < 8; ++e) {
#pragma unroll
      for (int o = 1; o < 16; o <<= 1) sm[e] += __shfl_xor(sm[e], o);
      lst[e] = lst[e] * sc[e] + sm[e];
    }
#pragma unroll
    for (int td = 0; td < 4; ++td)
#pragma unroll
      for (int e = 0; e < 8; ++e) O[td][e] *= sc[e];

    // ---- O += P @ V (contraction over 64 keys = 2 x 32) ----
#pragma unroll
    for (int ks = 0; ks < 2; ++ks) {
      v16bf ap = load_frag_a(&Ps[(wv * 16) * 72 + ks * 32], 72, lane);
#pragma unroll
      for (int td = 0; td < 4; ++td) {
        v16bf bv = load_frag_b(&Vt[buf][(td * 16) * 72 + ks * 32], 72, lane);
        O[td] = wmma_bf16(ap, bv, O[td]);
      }
    }
    __syncthreads();  // compute(buf) done before buf is restaged
  }

  // ---- epilogue: normalize, write head-interleaved concat layout ----
  const int rh = (lane >> 4) * 8;
#pragma unroll
  for (int e = 0; e < 8; ++e) {
    int s = q0 + wv * 16 + e + rh;
    float inv = 1.0f / lst[e];
    int crow = h * 64 + (s >> 4);
    int ccol0 = (s & 15) * 64;
#pragma unroll
    for (int td = 0; td < 4; ++td) {
      int d = td * 16 + cn;
      Cout[((size_t)b * 1024 + crow) * 1024 + ccol0 + d] = f2bf(O[td][e] * inv);
    }
  }
}

// ---------------------------------------------------------------------------
// Launch
// ---------------------------------------------------------------------------
extern "C" void kernel_launch(void* const* d_in, const int* in_sizes, int n_in,
                              void* d_out, int out_size, void* d_ws,
                              size_t ws_size, hipStream_t stream) {
  const float* q  = (const float*)d_in[0];
  const float* k  = (const float*)d_in[1];
  const float* v  = (const float*)d_in[2];
  const int* mask = (const int*)d_in[3];
  const float* Wq = (const float*)d_in[4];
  const float* Wk = (const float*)d_in[5];
  const float* Wv = (const float*)d_in[6];
  const float* Wo = (const float*)d_in[7];

  const int M = 16 * 1024, N = 1024;
  const size_t NEL = (size_t)M * N;   // 16M elements
  const size_t WEL = (size_t)N * N;   // 1M elements
  unsigned short* Qbf = (unsigned short*)d_ws;
  unsigned short* Kbf = Qbf + NEL;
  unsigned short* Vbf = Kbf + NEL;
  unsigned short* Cbf = Vbf + NEL;
  unsigned short* Wqb = Cbf + NEL;
  unsigned short* Wkb = Wqb + WEL;
  unsigned short* Wvb = Wkb + WEL;
  unsigned short* Wob = Wvb + WEL;

  const int cg = (int)(WEL / (256 * 4));  // 1024 blocks
  cvt_bf16<<<cg, 256, 0, stream>>>(Wq, Wqb);
  cvt_bf16<<<cg, 256, 0, stream>>>(Wk, Wkb);
  cvt_bf16<<<cg, 256, 0, stream>>>(Wv, Wvb);
  cvt_bf16<<<cg, 256, 0, stream>>>(Wo, Wob);

  dim3 gg(N / 128, M / 128);  // (8, 128)
  // Q gets 1/sqrt(DH) = 1/8 folded in.
  gemm_xwt<false, true><<<gg, 256, 0, stream>>>(q, Wqb, Qbf, 0.125f);
  gemm_xwt<false, true><<<gg, 256, 0, stream>>>(k, Wkb, Kbf, 1.0f);
  gemm_xwt<false, true><<<gg, 256, 0, stream>>>(v, Wvb, Vbf, 1.0f);

  attn_kernel<<<2048, 256, 0, stream>>>(Qbf, Kbf, Vbf, mask, Cbf);

  gemm_xwt<true, false><<<gg, 256, 0, stream>>>(Cbf, Wob, d_out, 1.0f);
}
